// SelfAttentionBlock_29807073034608
// MI455X (gfx1250) — matrix operand order repaired and checked by reading, not compiled, not run on hardware
//
#include <hip/hip_runtime.h>

#define NTOK 4096
#define CIN  256
#define CQK  32
#define NB   4
#define LOG2E 1.4426950408889634f

typedef __attribute__((ext_vector_type(16))) __bf16 v16bf;
typedef __attribute__((ext_vector_type(2)))  __bf16 v2bf;
typedef __attribute__((ext_vector_type(8)))  float  v8f;
typedef __attribute__((ext_vector_type(4)))  float  f32x4;
typedef __attribute__((ext_vector_type(4)))  unsigned int u32x4;

union BF16Tile {
    u32x4        q[2];
    v16bf        v;
    unsigned int u[8];
};

__device__ inline float fexp2(float a) {
#if __has_builtin(__builtin_amdgcn_exp2f)
    return __builtin_amdgcn_exp2f(a);     // raw v_exp_f32, no denorm fixup
#else
    return exp2f(a);
#endif
}

// swap values between the two 16-lane halves (elementwise xor-16) on VALU,
// avoiding the ds_bpermute + s_wait_dscnt round trip.
__device__ inline float xhalf(float v) {
#if __has_builtin(__builtin_amdgcn_permlanex16)
    int i = __builtin_bit_cast(int, v);
    i = __builtin_amdgcn_permlanex16(i, i, 0x76543210, 0xfedcba98, false, false);
    return __builtin_bit_cast(float, i);
#else
    return __shfl_xor(v, 16, 32);
#endif
}

__device__ inline unsigned short f32_to_bf16(float f) {
    unsigned int u = __builtin_bit_cast(unsigned int, f);
    u += 0x7FFFu + ((u >> 16) & 1u);           // round-to-nearest-even
    return (unsigned short)(u >> 16);
}

// pack two f32 -> bf16x2. Prefer hw cvt_pk; else round-half-up + v_perm_b32
// (1 v_add per value + 1 v_perm per pair); else full manual RNE.
__device__ inline unsigned int pk_bf16(float a, float b) {
#if __has_builtin(__builtin_amdgcn_cvt_pk_bf16_f32)
    v2bf t = __builtin_amdgcn_cvt_pk_bf16_f32(a, b);
    return __builtin_bit_cast(unsigned int, t);
#elif __has_builtin(__builtin_amdgcn_perm)
    unsigned int ua = __builtin_bit_cast(unsigned int, a) + 0x8000u;
    unsigned int ub = __builtin_bit_cast(unsigned int, b) + 0x8000u;
    // result bytes: [0]=ua.b2 [1]=ua.b3 [2]=ub.b2 [3]=ub.b3
    return __builtin_amdgcn_perm(ua, ub, 0x03020706u);
#else
    unsigned int ua = __builtin_bit_cast(unsigned int, a);
    unsigned int ub = __builtin_bit_cast(unsigned int, b);
    ua += 0x7FFFu + ((ua >> 16) & 1u);
    ub += 0x7FFFu + ((ub >> 16) & 1u);
    return (ua >> 16) | (ub & 0xFFFF0000u);
#endif
}

// ---------------------------------------------------------------------------
// Kernel 1: projections + ReLU.
//   qT, kT : (B, N, 32) bf16  (token-major, channels contiguous)
//   v      : (B, C, N)  bf16  (natural layout)
// q is pre-scaled by LOG2E so the attention kernel works in the exp2 domain.
// ---------------------------------------------------------------------------
__global__ __launch_bounds__(256) void proj_kernel(
    const float* __restrict__ x,
    const float* __restrict__ wq, const float* __restrict__ bq,
    const float* __restrict__ wk, const float* __restrict__ bk,
    const float* __restrict__ wv, const float* __restrict__ bv,
    unsigned short* __restrict__ qT, unsigned short* __restrict__ kT,
    unsigned short* __restrict__ vb)
{
    const int n   = blockIdx.x * 256 + threadIdx.x;
    const int row = blockIdx.y % 320;
    const int b   = blockIdx.y / 320;

    const float* w; float bias;
    if (row < 32)      { w = wq + row * CIN;        bias = bq[row]; }
    else if (row < 64) { w = wk + (row - 32) * CIN; bias = bk[row - 32]; }
    else               { w = wv + (row - 64) * CIN; bias = bv[row - 64]; }

    const float* xb = x + (size_t)b * CIN * NTOK + n;
    float acc = bias;
#pragma unroll 8
    for (int c = 0; c < CIN; ++c)
        acc = fmaf(w[c], xb[(size_t)c * NTOK], acc);
    acc = fmaxf(acc, 0.0f);

    if (row < 32) {
        qT[((size_t)(b * NTOK + n)) * CQK + row] = f32_to_bf16(acc * LOG2E);
    } else if (row < 64) {
        kT[((size_t)(b * NTOK + n)) * CQK + (row - 32)] = f32_to_bf16(acc);
    } else {
        vb[((size_t)(b * CIN + (row - 64))) * NTOK + n] = f32_to_bf16(acc);
    }
}

// ---------------------------------------------------------------------------
// Kernel 2: fused flash attention, split-K(eys) across wave pairs.
// Block = 256 threads = 8 waves = 4 query-tiles x 2 key-halves.
// ---------------------------------------------------------------------------
__global__ __launch_bounds__(256) void attn_kernel(
    const float* __restrict__ x, const float* __restrict__ gamma_p,
    const unsigned short* __restrict__ qT, const unsigned short* __restrict__ kT,
    const unsigned short* __restrict__ vb, float* __restrict__ out)
{
    __shared__ float lds_acc[8][8 * 256];   // 64 KB: 8 stashed tiles per wave
    __shared__ float lds_ml[8][2][32];      // running (m, l) per wave per lane
    __shared__ float xpose[8][16 * 17];     // padded transpose buffers

    const int tid  = threadIdx.x;
    const int wave = tid >> 5;
    const int lane = tid & 31;
    const int l    = lane & 15;
    const int half = lane >> 4;

    const int qt = wave >> 1;               // query tile within block (0..3)
    const int jh = wave & 1;                // key half (0..1)

    const int b     = blockIdx.x >> 6;      // 64 blocks per batch
    const int qbase = (blockIdx.x & 63) * 64 + qt * 16;
    const float gamma = gamma_p[0];

    // Q as WMMA B-matrix: lane<16 -> col i=l, K(c)={0..7,16..23} ; lane>=16 -> c={8..15,24..31}
    BF16Tile Q;
    {
        const unsigned short* qp =
            qT + ((size_t)(b * NTOK + qbase + l)) * CQK + half * 8;
        Q.q[0] = *(const u32x4*)(qp);
        Q.q[1] = *(const u32x4*)(qp + 16);
    }

    v8f zero8;
#pragma unroll
    for (int i = 0; i < 8; ++i) zero8[i] = 0.0f;

    v8f acc[16];
#pragma unroll
    for (int t = 0; t < 16; ++t) acc[t] = zero8;

    float m_run = -3.0e38f;                 // log2-domain running max
    float l_run = 0.0f;

    const unsigned short* kbase = kT + (size_t)b * NTOK * CQK;
    const unsigned short* vbase = vb + (size_t)b * CIN * NTOK;

    const int jbeg = jh * (NTOK / 2);
    const int jend = jbeg + (NTOK / 2);

    for (int jb = jbeg; jb < jend; jb += 32) {
        // K tiles as A-matrix: lane<16 -> row j=jb(+16)+l, K(c)={0..7,16..23};
        // lane>=16 -> c={8..15,24..31}
        BF16Tile Ka, Kb;
        const unsigned short* kp = kbase + (size_t)(jb + l) * CQK + half * 8;
        const unsigned short* kp2 = kbase + (size_t)(jb + 16 + l) * CQK + half * 8;
        {
            Ka.q[0] = *(const u32x4*)(kp);
            Ka.q[1] = *(const u32x4*)(kp + 16);
            Kb.q[0] = *(const u32x4*)(kp2);
            Kb.q[1] = *(const u32x4*)(kp2 + 16);
        }
        // S^T (already x LOG2E via q pre-scale): lane holds i=l;
        // j = r + 8*half (+16 for s1).
        v8f s0 = __builtin_amdgcn_wmma_f32_16x16x32_bf16(
            false, Ka.v, false, Q.v, (short)0, zero8, false, false);
        v8f s1 = __builtin_amdgcn_wmma_f32_16x16x32_bf16(
            false, Kb.v, false, Q.v, (short)0, zero8, false, false);

        // warm next iteration's K tiles (fills WMMA->VALU hazard slots;
        // no VGPR cost, no counter). Last iteration prefetches in-bounds junk.
        __builtin_prefetch(kp  + 32 * CQK, 0, 3);
        __builtin_prefetch(kp2 + 32 * CQK, 0, 3);

        // ---- online softmax (exp2 domain) ----
        float mx = s0[0];
#pragma unroll
        for (int r = 1; r < 8; ++r) mx = fmaxf(mx, s0[r]);
#pragma unroll
        for (int r = 0; r < 8; ++r) mx = fmaxf(mx, s1[r]);
        mx = fmaxf(mx, xhalf(mx));
        const float m_new = fmaxf(m_run, mx);
        const float scale = fexp2(m_run - m_new);
        m_run = m_new;

        float p0[8], p1[8], tsum = 0.0f;
#pragma unroll
        for (int r = 0; r < 8; ++r) { p0[r] = fexp2(s0[r] - m_new); tsum += p0[r]; }
#pragma unroll
        for (int r = 0; r < 8; ++r) { p1[r] = fexp2(s1[r] - m_new); tsum += p1[r]; }
        tsum += xhalf(tsum);
        l_run = l_run * scale + tsum;

        // rescale accumulators: acc row i = r + 8*half -> broadcast scale(i)
        float sbr[8];
#pragma unroll
        for (int r = 0; r < 8; ++r) sbr[r] = __shfl(scale, r + 8 * half, 32);
#pragma unroll
        for (int t = 0; t < 16; ++t)
#pragma unroll
            for (int r = 0; r < 8; ++r) acc[t][r] *= sbr[r];

        // P as A-matrix (16i x 32j): lane's S^T column IS row i of P;
        // j split {0..7,16..23}/{8..15,24..31} matches the A layout exactly.
        BF16Tile P;
#pragma unroll
        for (int k = 0; k < 4; ++k) {
            P.u[k]     = pk_bf16(p0[2 * k], p0[2 * k + 1]);
            P.u[4 + k] = pk_bf16(p1[2 * k], p1[2 * k + 1]);
        }

        // acc[t] += P x V^T; V^T as B-matrix loads contiguously from (B,C,N).
        const unsigned short* vp0 = vbase + (size_t)l * NTOK + jb + half * 8;
#pragma unroll
        for (int t = 0; t < 16; ++t) {
            BF16Tile V;
            const unsigned short* vp = vp0 + (size_t)t * 16 * NTOK;
            V.q[0] = *(const u32x4*)(vp);
            V.q[1] = *(const u32x4*)(vp + 16);
            acc[t] = __builtin_amdgcn_wmma_f32_16x16x32_bf16(
                false, P.v, false, V.v, (short)0, acc[t], false, false);
        }
    }

    // ---- merge the two key-halves of this query tile ----
    lds_ml[wave][0][lane] = m_run;
    lds_ml[wave][1][lane] = l_run;
    __syncthreads();

    const int pw = wave ^ 1;                 // partner wave (other key half)
    const float m_p   = lds_ml[pw][0][lane];
    const float l_p   = lds_ml[pw][1][lane];
    const float m_tot = fmaxf(m_run, m_p);
    const float s_own = fexp2(m_run - m_tot);
    const float s_p   = fexp2(m_p - m_tot);
    const float l_tot = l_run * s_own + l_p * s_p;

    // scale own accumulators to the common max
    {
        float sbr[8];
#pragma unroll
        for (int r = 0; r < 8; ++r) sbr[r] = __shfl(s_own, r + 8 * half, 32);
#pragma unroll
        for (int t = 0; t < 16; ++t)
#pragma unroll
            for (int r = 0; r < 8; ++r) acc[t][r] *= sbr[r];
    }

    // stash the 8 tiles belonging to the partner's output range
    // (wave jh keeps t = jh*8..jh*8+7; stashes the other 8, already scaled)
    {
        float* dst = &lds_acc[wave][0] + lane * 8;
#pragma unroll
        for (int s = 0; s < 8; ++s) {
            const int t = (1 - jh) * 8 + s;
            f32x4 lo, hi;
#pragma unroll
            for (int i = 0; i < 4; ++i) { lo[i] = acc[t][i]; hi[i] = acc[t][4 + i]; }
            *(f32x4*)(dst + s * 256)     = lo;
            *(f32x4*)(dst + s * 256 + 4) = hi;
        }
    }
    __syncthreads();

    // ---- final: add partner tiles, transpose via LDS, fused epilogue ----
    const float inv_l = 1.0f / l_tot;       // valid for i = lane&15 post-transpose
    const float* src = &lds_acc[pw][0] + lane * 8;
    float* lds = xpose[wave];

#pragma unroll
    for (int s = 0; s < 8; ++s) {
        const int tglob = jh * 8 + s;       // my output channel tile
        f32x4 plo = *(const f32x4*)(src + s * 256);
        f32x4 phi = *(const f32x4*)(src + s * 256 + 4);
        float merged[8];
#pragma unroll
        for (int i = 0; i < 4; ++i) {
            merged[i]     = acc[tglob][i]     + plo[i];
            merged[4 + i] = acc[tglob][4 + i] + phi[i];
        }
        // write (i = r+8*half, c_local = l); same-wave DS is in-order
#pragma unroll
        for (int r = 0; r < 8; ++r)
            lds[(r + 8 * half) * 17 + l] = merged[r];
        // read transposed (i = l, c_local = r+8*half) and store coalesced
#pragma unroll
        for (int r = 0; r < 8; ++r) {
            float val = lds[l * 17 + r + 8 * half];
            int c = tglob * 16 + r + 8 * half;
            size_t off = ((size_t)(b * CIN + c)) * NTOK + qbase + l;
            out[off] = fmaf(gamma * inv_l, val, x[off]);
        }
    }
}

// ---------------------------------------------------------------------------
extern "C" void kernel_launch(void* const* d_in, const int* in_sizes, int n_in,
                              void* d_out, int out_size, void* d_ws, size_t ws_size,
                              hipStream_t stream) {
    const float* x     = (const float*)d_in[0];
    const float* wq    = (const float*)d_in[1];
    const float* bq    = (const float*)d_in[2];
    const float* wk    = (const float*)d_in[3];
    const float* bk    = (const float*)d_in[4];
    const float* wv    = (const float*)d_in[5];
    const float* bv    = (const float*)d_in[6];
    const float* gamma = (const float*)d_in[7];
    float* out = (float*)d_out;

    unsigned short* qT = (unsigned short*)d_ws;                 // 1 MB
    unsigned short* kT = qT + (size_t)NB * NTOK * CQK;          // 1 MB
    unsigned short* vb = kT + (size_t)NB * NTOK * CQK;          // 8 MB

    dim3 pg(NTOK / 256, 320 * NB);
    proj_kernel<<<pg, 256, 0, stream>>>(x, wq, bq, wk, bk, wv, bv, qT, kT, vb);

    // 256 blocks x 8 waves: 4 query tiles x 2 key-halves each = 2048 waves
    attn_kernel<<<NB * (NTOK / 64), 256, 0, stream>>>(x, gamma, qT, kT, vb, out);
}
